// GraphConvolution_5248450035900
// MI455X (gfx1250) — matrix operand order repaired and checked by reading, not compiled, not run on hardware
//
#include <hip/hip_runtime.h>
#include <stdint.h>

// Problem constants (fixed by the reference harness)
#define NROWS   2048    // N
#define KDIM    1024    // D_IN
#define DOUT    8192    // D_OUT
#define EDGES   131072  // E

typedef __attribute__((ext_vector_type(16))) __bf16 v16bf;
typedef __attribute__((ext_vector_type(8)))  float  v8f;
typedef unsigned short u16;

struct __align__(16) U128 { unsigned int x[4]; };
struct __align__(16) B256 { U128 a, b; };

__device__ __forceinline__ u16 f32_to_bf16_rne(float f) {
  unsigned int u = __float_as_uint(f);
  u += 0x7fffu + ((u >> 16) & 1u);          // round-to-nearest-even
  return (u16)(u >> 16);
}
__device__ __forceinline__ float bf16_to_f32(u16 h) {
  return __uint_as_float(((unsigned int)h) << 16);
}

// Async global->LDS copy, 16B per lane, ASYNCcnt-tracked (CDNA5 §10 async ops).
// LDS operand: generic pointer low 32 bits == LDS byte offset (aperture rule).
__device__ __forceinline__ void async_copy_b128(u16* lds, const u16* g) {
  asm volatile("global_load_async_to_lds_b128 %0, %1, off"
               :: "v"((unsigned int)(uintptr_t)lds),
                  "v"((unsigned long long)(uintptr_t)g)
               : "memory");
}

#define WMMA_BF16(acc, a, b)                                                   \
  acc = __builtin_amdgcn_wmma_f32_16x16x32_bf16(false, a, false, b, (short)0,  \
                                                acc, false, false)

// A operand (16x32 bf16, ISA 7.12.2): 16B at p, 16B at p+32B
__device__ __forceinline__ v16bf ld_a(const u16* p) {
  B256 t; t.a = *(const U128*)p; t.b = *(const U128*)(p + 16);
  return __builtin_bit_cast(v16bf, t);
}
// B operand (32x16 bf16): 16B at p, 16B at p+32B
__device__ __forceinline__ v16bf ld_b(const u16* p) {
  B256 t; t.a = *(const U128*)p; t.b = *(const U128*)(p + 16);
  return __builtin_bit_cast(v16bf, t);
}

// ---------------------------------------------------------------------------
// Prepass 1: split input [N][K] f32 -> hi/lo bf16 [N][K]
// ---------------------------------------------------------------------------
__global__ __launch_bounds__(256) void split_bf16(const float* __restrict__ src,
                                                  u16* __restrict__ hi,
                                                  u16* __restrict__ lo, int n) {
  int i = blockIdx.x * 256 + threadIdx.x;
  if (i < n) {
    float x = src[i];
    u16 h = f32_to_bf16_rne(x);
    hi[i] = h;
    lo[i] = f32_to_bf16_rne(x - bf16_to_f32(h));
  }
}

// ---------------------------------------------------------------------------
// Prepass 2: weight [K][DOUT] f32 -> transposed hi/lo bf16 [DOUT][K]
// ---------------------------------------------------------------------------
__global__ __launch_bounds__(256) void wt_transpose_split(const float* __restrict__ W,
                                                          u16* __restrict__ hiT,
                                                          u16* __restrict__ loT) {
  __shared__ float t[32][33];
  const int n0 = blockIdx.x * 32;
  const int k0 = blockIdx.y * 32;
  const int tx = threadIdx.x, ty = threadIdx.y;
#pragma unroll
  for (int i = 0; i < 4; ++i)
    t[ty + i * 8][tx] = W[(size_t)(k0 + ty + i * 8) * DOUT + n0 + tx];
  __syncthreads();
#pragma unroll
  for (int i = 0; i < 4; ++i) {
    float x = t[tx][ty + i * 8];
    size_t o = (size_t)(n0 + ty + i * 8) * KDIM + k0 + tx;
    u16 h = f32_to_bf16_rne(x);
    hiT[o] = h;
    loT[o] = f32_to_bf16_rne(x - bf16_to_f32(h));
  }
}

// ---------------------------------------------------------------------------
// GEMM: support_T[DOUT][N] = W^T @ input^T, bf16x3 error-compensated WMMA.
// Block = 128 threads = 4 wave32 (2x2), block tile 64x64, wave tile 32x32
// (2x2 of 16x16 accumulators -> 12 WMMA per 16 LDS b128 per K-chunk of 32).
// Double-buffered LDS filled with GLOBAL_LOAD_ASYNC_TO_LDS_B128 (ASYNCcnt).
// ---------------------------------------------------------------------------
#define BM 64
#define BN 64
#define BK 32
#define LSTR 56   // LDS row stride in bf16 (112B: 16B-aligned, bank-spread)

__global__ __launch_bounds__(128) void gemm_bf16x3(
    const u16* __restrict__ Whi, const u16* __restrict__ Wlo,   // [DOUT][KDIM]
    const u16* __restrict__ Xhi, const u16* __restrict__ Xlo,   // [NROWS][KDIM]
    float* __restrict__ St)                                     // [DOUT][NROWS]
{
  __shared__ u16 sAh[2][BM * LSTR], sAl[2][BM * LSTR];
  __shared__ u16 sBh[2][BN * LSTR], sBl[2][BN * LSTR];

  const int tid  = threadIdx.x;
  const int bm   = blockIdx.y * BM;       // dout block row
  const int bn   = blockIdx.x * BN;       // n block col
  const int srow = tid >> 1;              // staging row 0..63
  const int scol = (tid & 1) * 16;        // staging col (u16 units), 32B half-row

  const int lane = tid & 31;
  const int wave = tid >> 5;
  const int wm   = (wave & 1) * 32;       // wave M offset (2 m-subtiles)
  const int wn   = (wave >> 1) * 32;      // wave N offset (2 n-subtiles)
  const int m    = lane & 15;
  const int half = lane >> 4;

  const size_t gA = (size_t)(bm + srow) * KDIM + scol;
  const size_t gB = (size_t)(bn + srow) * KDIM + scol;
  const int    sO = srow * LSTR + scol;

  // Issue one K-chunk's staging: 8 async b128 per thread (ASYNCcnt += 8/wave).
  auto stage = [&](int kc, int buf) {
    async_copy_b128(&sAh[buf][sO],     Whi + gA + kc);
    async_copy_b128(&sAh[buf][sO + 8], Whi + gA + kc + 8);
    async_copy_b128(&sAl[buf][sO],     Wlo + gA + kc);
    async_copy_b128(&sAl[buf][sO + 8], Wlo + gA + kc + 8);
    async_copy_b128(&sBh[buf][sO],     Xhi + gB + kc);
    async_copy_b128(&sBh[buf][sO + 8], Xhi + gB + kc + 8);
    async_copy_b128(&sBl[buf][sO],     Xlo + gB + kc);
    async_copy_b128(&sBl[buf][sO + 8], Xlo + gB + kc + 8);
  };

  v8f acc00 = {0, 0, 0, 0, 0, 0, 0, 0};
  v8f acc01 = {0, 0, 0, 0, 0, 0, 0, 0};
  v8f acc10 = {0, 0, 0, 0, 0, 0, 0, 0};
  v8f acc11 = {0, 0, 0, 0, 0, 0, 0, 0};

  stage(0, 0);                                        // prologue: chunk 0

  for (int kc = 0; kc < KDIM; kc += BK) {
    const int buf = (kc >> 5) & 1;
    if (kc + BK < KDIM) {
      stage(kc + BK, buf ^ 1);                        // prefetch next chunk
      // async loads complete in order: <=8 outstanding => current chunk landed
      asm volatile("s_wait_asynccnt 0x8" ::: "memory");
    } else {
      asm volatile("s_wait_asynccnt 0x0" ::: "memory");
    }
    __syncthreads();

    const u16* pA = &sAh[buf][(wm + m) * LSTR + half * 8];
    const u16* qA = &sAl[buf][(wm + m) * LSTR + half * 8];
    const u16* pB = &sBh[buf][(wn + m) * LSTR + half * 8];
    const u16* qB = &sBl[buf][(wn + m) * LSTR + half * 8];

    v16bf ah0 = ld_a(pA);
    v16bf al0 = ld_a(qA);
    v16bf ah1 = ld_a(pA + 16 * LSTR);
    v16bf al1 = ld_a(qA + 16 * LSTR);
    v16bf bh0 = ld_b(pB);
    v16bf bl0 = ld_b(qB);
    v16bf bh1 = ld_b(pB + 16 * LSTR);
    v16bf bl1 = ld_b(qB + 16 * LSTR);

    // bf16x3: hi*hi + hi*lo + lo*hi, f32 accumulate; 2x2 operand reuse
    WMMA_BF16(acc00, ah0, bh0); WMMA_BF16(acc00, ah0, bl0); WMMA_BF16(acc00, al0, bh0);
    WMMA_BF16(acc01, ah0, bh1); WMMA_BF16(acc01, ah0, bl1); WMMA_BF16(acc01, al0, bh1);
    WMMA_BF16(acc10, ah1, bh0); WMMA_BF16(acc10, ah1, bl0); WMMA_BF16(acc10, al1, bh0);
    WMMA_BF16(acc11, ah1, bh1); WMMA_BF16(acc11, ah1, bl1); WMMA_BF16(acc11, al1, bh1);

    __syncthreads();   // all waves done reading buf before it is refilled
  }

  // C layout: lanes 0-15 N=lane, M=i; lanes 16-31 N=lane-16, M=8+i
  float* d0 = St + (size_t)(bm + wm + half * 8) * NROWS + (bn + wn + m);
  float* d1 = d0 + (size_t)16 * NROWS;
#pragma unroll
  for (int i = 0; i < 8; ++i) {
    d0[(size_t)i * NROWS]      = acc00[i];
    d0[(size_t)i * NROWS + 16] = acc01[i];
    d1[(size_t)i * NROWS]      = acc10[i];
    d1[(size_t)i * NROWS + 16] = acc11[i];
  }
}

// ---------------------------------------------------------------------------
// SpMM scatter in transposed space: one wave per edge; coalesced 8KB streams.
// agg_T[r][:] += val * support_T[c][:]   (agg_T fits in 192MB L2 -> L2 atomics)
// ---------------------------------------------------------------------------
__global__ __launch_bounds__(256) void spmm_scatter(const int* __restrict__ rows,
                                                    const int* __restrict__ cols,
                                                    const float* __restrict__ vals,
                                                    const float* __restrict__ St,
                                                    float* __restrict__ Gt) {
  const int e    = blockIdx.x * 8 + (threadIdx.x >> 5);
  const int lane = threadIdx.x & 31;
  const int r = rows[e], c = cols[e];
  const float v = vals[e];
  const float* src = St + (size_t)c * NROWS;
  float*       dst = Gt + (size_t)r * NROWS;
#pragma unroll 4
  for (int n = lane; n < NROWS; n += 32)
    atomicAdd(dst + n, v * src[n]);
}

// ---------------------------------------------------------------------------
// Epilogue: out[n][r] = agg_T[r][n] + bias[r]  (LDS-tiled transpose)
// ---------------------------------------------------------------------------
__global__ __launch_bounds__(256) void transpose_add_bias(const float* __restrict__ Gt,
                                                          const float* __restrict__ bias,
                                                          float* __restrict__ out) {
  __shared__ float t[32][33];
  const int r0 = blockIdx.x * 32;
  const int n0 = blockIdx.y * 32;
  const int tx = threadIdx.x, ty = threadIdx.y;
#pragma unroll
  for (int i = 0; i < 4; ++i)
    t[ty + i * 8][tx] = Gt[(size_t)(r0 + ty + i * 8) * NROWS + n0 + tx];
  __syncthreads();
  const float b = bias[r0 + tx];
#pragma unroll
  for (int i = 0; i < 4; ++i)
    out[(size_t)(n0 + ty + i * 8) * DOUT + r0 + tx] = t[tx][ty + i * 8] + b;
}

// ---------------------------------------------------------------------------
extern "C" void kernel_launch(void* const* d_in, const int* in_sizes, int n_in,
                              void* d_out, int out_size, void* d_ws, size_t ws_size,
                              hipStream_t stream) {
  (void)in_sizes; (void)n_in; (void)out_size; (void)ws_size;
  const float* input  = (const float*)d_in[0];
  const float* weight = (const float*)d_in[1];
  const float* bias   = (const float*)d_in[2];
  const int*   arows  = (const int*)d_in[3];
  const int*   acols  = (const int*)d_in[4];
  const float* avals  = (const float*)d_in[5];
  float* out = (float*)d_out;

  // Workspace layout (total 168 MB)
  char* ws = (char*)d_ws;
  const size_t szSt = (size_t)DOUT * NROWS * sizeof(float);   // 64 MB support_T
  const size_t szGt = szSt;                                   // 64 MB agg_T
  const size_t szX  = (size_t)NROWS * KDIM * sizeof(u16);     //  4 MB each (input hi/lo)
  const size_t szW  = (size_t)DOUT * KDIM * sizeof(u16);      // 16 MB each (W^T hi/lo)
  float* St  = (float*)(ws);
  float* Gt  = (float*)(ws + szSt);
  u16*   Xhi = (u16*)(ws + szSt + szGt);
  u16*   Xlo = (u16*)(ws + szSt + szGt + szX);
  u16*   Whi = (u16*)(ws + szSt + szGt + 2 * szX);
  u16*   Wlo = (u16*)(ws + szSt + szGt + 2 * szX + szW);

  split_bf16<<<(NROWS * KDIM) / 256, 256, 0, stream>>>(input, Xhi, Xlo, NROWS * KDIM);
  wt_transpose_split<<<dim3(DOUT / 32, KDIM / 32), dim3(32, 8), 0, stream>>>(weight, Whi, Wlo);
  gemm_bf16x3<<<dim3(NROWS / BN, DOUT / BM), 128, 0, stream>>>(Whi, Wlo, Xhi, Xlo, St);
  hipMemsetAsync(Gt, 0, szGt, stream);
  spmm_scatter<<<EDGES / 8, 256, 0, stream>>>(arows, acols, avals, St, Gt);
  transpose_add_bias<<<dim3(DOUT / 32, NROWS / 32), dim3(32, 8), 0, stream>>>(Gt, bias, out);
}
